// GCN_graph_bn_23716809408544
// MI455X (gfx1250) — matrix operand order repaired and checked by reading, not compiled, not run on hardware
//
#include <hip/hip_runtime.h>
#include <hip/hip_bf16.h>
#include <math.h>

typedef __attribute__((ext_vector_type(16))) _Float16 v16h;
typedef __attribute__((ext_vector_type(8)))  float    v8f;

#define NNODES 50000
#define NEDGES 800000
#define NGRAPH 1024
#define DFEAT  128
#define D3F    64
#define KDIM   128      // all GEMM K dims are 128
#define KSTEPS 4        // 128 / 32
#define BN_EPS 1e-5f

__device__ __forceinline__ int clampi(int v, int hi) {
  return v < 0 ? 0 : (v >= hi ? hi - 1 : v);
}
__device__ __forceinline__ void atom_add_f(float* p, float v) {
  __hip_atomic_fetch_add(p, v, __ATOMIC_RELAXED, __HIP_MEMORY_SCOPE_AGENT);
}
__device__ __forceinline__ void atom_max_u(unsigned* p, unsigned v) {
  __hip_atomic_fetch_max(p, v, __ATOMIC_RELAXED, __HIP_MEMORY_SCOPE_AGENT);
}

// ---------------- degree / normalization ----------------
__global__ void deg_init_kernel(float* deg) {
  int i = blockIdx.x * blockDim.x + threadIdx.x;
  if (i < NNODES) deg[i] = 1.0f;  // self-loop
}
__global__ void deg_count_kernel(const int* __restrict__ col, float* __restrict__ deg) {
  int e = blockIdx.x * blockDim.x + threadIdx.x;
  if (e < NEDGES) atom_add_f(&deg[clampi(col[e], NNODES)], 1.0f);
}
__global__ void dinv_kernel(float* deg) {
  int i = blockIdx.x * blockDim.x + threadIdx.x;
  if (i < NNODES) deg[i] = rsqrtf(deg[i]);
}

// ---------------- weight pre-pack: f32 [K,Nn] -> f16 WMMA B-fragment order ----
// Wp laid out as [ct][ks][lane][16 halves]; lane<16: N=lane, K=k0..k0+15;
// lane>=16: N=lane-16, K=k0+16..k0+31  (contiguous k order, matching this file's A fragments)
__global__ void prepack_w(const float* __restrict__ W, _Float16* __restrict__ Wp, int Nn) {
  int t = blockIdx.x * blockDim.x + threadIdx.x;
  if (t >= KDIM * Nn) return;
  int i    = t & 15;
  int lane = (t >> 4) & 31;
  int rest = t >> 9;
  int ks   = rest % KSTEPS;
  int ct   = rest / KSTEPS;
  int n  = lane & 15;
  int ko = (lane >> 4) << 4;
  int k  = ks * 32 + ko + i;
  Wp[t] = (_Float16)W[(size_t)k * Nn + ct * 16 + n];
}

// ---------------- WMMA GEMM: C[M,NN] = A[M,128] @ W[128,NN] ----------------
// block = 256 threads (8 waves). NN=128: 8 col tiles, 16 rows/block.
// NN=64: 4 col tiles, 2 row tiles -> 32 rows/block. A staged in LDS as f16.
template <int NN>
__global__ void gemm_wmma_opt(const float* __restrict__ A,
                              const _Float16* __restrict__ Wp,
                              float* __restrict__ C) {
  constexpr int COLT = NN / 16;        // col tiles per block (8 or 4)
  constexpr int ROWT = 8 / COLT;       // row tiles per block (1 or 2)
  constexpr int R    = 16 * ROWT;      // rows per block
  __shared__ _Float16 As[R * KDIM];

  const int tid  = threadIdx.x;
  const int lane = tid & 31;
  const int wave = tid >> 5;
  const int rt   = wave / COLT;
  const int ct   = wave % COLT;
  const int row0 = blockIdx.x * R;

  // cooperative A tile load + f32->f16 convert (coalesced)
  constexpr int ELEMS = R * KDIM / 256;
  const float* Ab = A + (size_t)row0 * KDIM;
#pragma unroll
  for (int i = 0; i < ELEMS; ++i) {
    int idx = tid + i * 256;
    As[idx] = (_Float16)Ab[idx];
  }
  __syncthreads();

  const int m  = lane & 15;
  const int kb = (lane >> 4) << 4;     // A frag: lanes 16-31 hold K+16..K+31
  const _Float16* Arow = As + (size_t)(rt * 16 + m) * KDIM;
  const v16h* Bf = (const v16h*)Wp + (size_t)ct * KSTEPS * 32 + lane;

  v8f acc = {};
#pragma unroll
  for (int ks = 0; ks < KSTEPS; ++ks) {
    v16h a;
    const _Float16* ap = Arow + ks * 32;
#pragma unroll
    for (int i = 0; i < 8; ++i) a[i]     = ap[kb + i];
#pragma unroll
    for (int i = 0; i < 8; ++i) a[8 + i] = ap[kb + 8 + i];
    v16h b = Bf[ks * 32];               // one contiguous 32B fragment load
    acc = __builtin_amdgcn_wmma_f32_16x16x32_f16(false, a, false, b,
                                                 (short)0, acc, false, false);
  }

  const int mo = (lane >> 4) << 3;      // C/D: lanes 16-31 hold M+8
  const int n  = lane & 15;
#pragma unroll
  for (int r = 0; r < 8; ++r)
    C[(size_t)(row0 + rt * 16 + mo + r) * NN + ct * 16 + n] = acc[r];
}

// ---------------- GCN aggregation ----------------
__global__ void selfloop_init(const float* __restrict__ XW, const float* __restrict__ dinv,
                              float* __restrict__ out) {
  size_t i = (size_t)blockIdx.x * blockDim.x + threadIdx.x;
  if (i >= (size_t)NNODES * DFEAT) return;
  int node = (int)(i >> 7);
  float d = dinv[node];
  out[i] = XW[i] * d * d;
}
__global__ void scatter_edges(const float* __restrict__ XW, const int* __restrict__ row,
                              const int* __restrict__ col, const float* __restrict__ dinv,
                              float* __restrict__ out) {
  size_t t = (size_t)blockIdx.x * blockDim.x + threadIdx.x;
  size_t e = t >> 5;
  if (e >= NEDGES) return;
  int f = (int)(t & 31) << 2;
  int r = clampi(row[e], NNODES);
  int c = clampi(col[e], NNODES);
  float nrm = dinv[r] * dinv[c];
  const float4 v = *(const float4*)(XW + ((size_t)r << 7) + f);
  float* o = out + ((size_t)c << 7) + f;
  atom_add_f(o + 0, v.x * nrm);
  atom_add_f(o + 1, v.y * nrm);
  atom_add_f(o + 2, v.z * nrm);
  atom_add_f(o + 3, v.w * nrm);
}

// ---------------- BatchNorm (training stats) + ReLU ----------------
__global__ void zero_f32(float* p, int n) {
  int i = blockIdx.x * blockDim.x + threadIdx.x;
  if (i < n) p[i] = 0.0f;
}
__global__ void bn_stats(const float* __restrict__ h, float* __restrict__ stats,
                         int M, int C) {
  int c = threadIdx.x % C;
  int rstep = blockDim.x / C;
  int r0 = blockIdx.x * 256 + threadIdx.x / C;
  int rend = min(M, (blockIdx.x + 1) * 256);
  float s = 0.0f, s2 = 0.0f;
  for (int r = r0; r < rend; r += rstep) {
    float v = h[(size_t)r * C + c];
    s += v; s2 += v * v;
  }
  atom_add_f(&stats[c], s);
  atom_add_f(&stats[C + c], s2);
}
__global__ void bn_apply_relu(float* __restrict__ h, const float* __restrict__ stats,
                              const float* __restrict__ g, const float* __restrict__ be,
                              int M, int C) {
  size_t i = (size_t)blockIdx.x * blockDim.x + threadIdx.x;
  if (i >= (size_t)M * C) return;
  int c = (int)(i % C);
  float inv_m = 1.0f / (float)M;
  float mu = stats[c] * inv_m;
  float var = stats[C + c] * inv_m - mu * mu;
  float v = (h[i] - mu) * rsqrtf(var + BN_EPS) * g[c] + be[c];
  h[i] = v > 0.0f ? v : 0.0f;
}

// ---------------- global_max_pool (values >= 0 after ReLU) ----------------
__global__ void pool_max(const float* __restrict__ h, const int* __restrict__ batch,
                         unsigned* __restrict__ pooled) {
  size_t t = (size_t)blockIdx.x * blockDim.x + threadIdx.x;
  size_t node = t >> 5;
  if (node >= NNODES) return;
  int f = (int)(t & 31) << 2;
  int gid = clampi(batch[node], NGRAPH);
  const float4 v = *(const float4*)(h + (node << 7) + f);
  unsigned* o = pooled + ((size_t)gid << 7) + f;
  atom_max_u(o + 0, __float_as_uint(v.x));
  atom_max_u(o + 1, __float_as_uint(v.y));
  atom_max_u(o + 2, __float_as_uint(v.z));
  atom_max_u(o + 3, __float_as_uint(v.w));
}

// ---------------- head: logits + log_softmax ----------------
__global__ void head_kernel(const float* __restrict__ q, const float* __restrict__ Wl2,
                            const float* __restrict__ bl2, float* __restrict__ out) {
  int g = blockIdx.x * blockDim.x + threadIdx.x;
  if (g >= NGRAPH) return;
  float z0 = bl2[0], z1 = bl2[1];
  const float* qr = q + (size_t)g * D3F;
#pragma unroll 4
  for (int k = 0; k < D3F; ++k) {
    float v = qr[k];
    z0 += v * Wl2[k * 2 + 0];
    z1 += v * Wl2[k * 2 + 1];
  }
  float mx = fmaxf(z0, z1);
  float lse = mx + logf(expf(z0 - mx) + expf(z1 - mx));
  out[g * 2 + 0] = z0 - lse;
  out[g * 2 + 1] = z1 - lse;
}

// ---------------- launcher ----------------
static inline size_t align_up(size_t v, size_t a) { return (v + a - 1) & ~(a - 1); }

extern "C" void kernel_launch(void* const* d_in, const int* in_sizes, int n_in,
                              void* d_out, int out_size, void* d_ws, size_t ws_size,
                              hipStream_t stream) {
  (void)in_sizes; (void)n_in; (void)out_size; (void)ws_size;
  const float* x   = (const float*)d_in[0];
  const int*   ei  = (const int*)d_in[1];      // [2,E]: row=ei, col=ei+E
  const int*   bat = (const int*)d_in[2];
  const float* W1  = (const float*)d_in[3];
  const float* g1  = (const float*)d_in[5];
  const float* be1 = (const float*)d_in[6];
  const float* W2  = (const float*)d_in[7];
  const float* g2  = (const float*)d_in[9];
  const float* be2 = (const float*)d_in[10];
  const float* Wl1 = (const float*)d_in[11];
  const float* g3  = (const float*)d_in[13];
  const float* be3 = (const float*)d_in[14];
  const float* Wl2 = (const float*)d_in[15];
  const float* bl2 = (const float*)d_in[16];
  float* out = (float*)d_out;
  // b1/b2/bl1 are absorbed by the following BatchNorm layers (shift cancels)

  char* w = (char*)d_ws;
  float*     dinv   = (float*)w;     w += align_up((size_t)NNODES * 4, 256);
  float*     bufA   = (float*)w;     w += align_up((size_t)NNODES * DFEAT * 4, 256);
  float*     bufB   = (float*)w;     w += align_up((size_t)NNODES * DFEAT * 4, 256);
  float*     stats  = (float*)w;     w += align_up(2 * DFEAT * 4, 256);
  float*     pooled = (float*)w;     w += align_up((size_t)NGRAPH * DFEAT * 4, 256);
  float*     q      = (float*)w;     w += align_up((size_t)NGRAPH * D3F * 4, 256);
  _Float16*  Wp     = (_Float16*)w;  w += align_up((size_t)KDIM * DFEAT * 2, 256);

  const int TB = 256;
  const int ND = NNODES * DFEAT;

  // 1) deg -> dinv
  deg_init_kernel<<<(NNODES + TB - 1) / TB, TB, 0, stream>>>(dinv);
  deg_count_kernel<<<(NEDGES + TB - 1) / TB, TB, 0, stream>>>(ei + NEDGES, dinv);
  dinv_kernel<<<(NNODES + TB - 1) / TB, TB, 0, stream>>>(dinv);

  // 2) layer 1: XW -> aggregate -> BN+ReLU   (bufB = h1)
  prepack_w<<<(KDIM * DFEAT + TB - 1) / TB, TB, 0, stream>>>(W1, Wp, DFEAT);
  gemm_wmma_opt<DFEAT><<<NNODES / 16, 256, 0, stream>>>(x, Wp, bufA);
  selfloop_init<<<(ND + TB - 1) / TB, TB, 0, stream>>>(bufA, dinv, bufB);
  scatter_edges<<<((size_t)NEDGES * 32 + TB - 1) / TB, TB, 0, stream>>>(bufA, ei, ei + NEDGES, dinv, bufB);
  zero_f32<<<1, 2 * DFEAT, 0, stream>>>(stats, 2 * DFEAT);
  bn_stats<<<(NNODES + 255) / 256, TB, 0, stream>>>(bufB, stats, NNODES, DFEAT);
  bn_apply_relu<<<(ND + TB - 1) / TB, TB, 0, stream>>>(bufB, stats, g1, be1, NNODES, DFEAT);

  // 3) layer 2
  prepack_w<<<(KDIM * DFEAT + TB - 1) / TB, TB, 0, stream>>>(W2, Wp, DFEAT);
  gemm_wmma_opt<DFEAT><<<NNODES / 16, 256, 0, stream>>>(bufB, Wp, bufA);
  selfloop_init<<<(ND + TB - 1) / TB, TB, 0, stream>>>(bufA, dinv, bufB);
  scatter_edges<<<((size_t)NEDGES * 32 + TB - 1) / TB, TB, 0, stream>>>(bufA, ei, ei + NEDGES, dinv, bufB);
  zero_f32<<<1, 2 * DFEAT, 0, stream>>>(stats, 2 * DFEAT);
  bn_stats<<<(NNODES + 255) / 256, TB, 0, stream>>>(bufB, stats, NNODES, DFEAT);
  bn_apply_relu<<<(ND + TB - 1) / TB, TB, 0, stream>>>(bufB, stats, g2, be2, NNODES, DFEAT);

  // 4) global max pool (bit-max valid: post-ReLU values are >= 0)
  zero_f32<<<(NGRAPH * DFEAT + TB - 1) / TB, TB, 0, stream>>>(pooled, NGRAPH * DFEAT);
  pool_max<<<((size_t)NNODES * 32 + TB - 1) / TB, TB, 0, stream>>>(bufB, bat, (unsigned*)pooled);

  // 5) linear head: pooled @ Wl1 -> BN+ReLU -> logits + log_softmax
  prepack_w<<<(KDIM * D3F + TB - 1) / TB, TB, 0, stream>>>(Wl1, Wp, D3F);
  gemm_wmma_opt<D3F><<<NGRAPH / 32, 256, 0, stream>>>(pooled, Wp, q);
  zero_f32<<<1, 2 * DFEAT, 0, stream>>>(stats, 2 * DFEAT);
  bn_stats<<<(NGRAPH + 255) / 256, TB, 0, stream>>>(q, stats, NGRAPH, D3F);
  bn_apply_relu<<<(NGRAPH * D3F + TB - 1) / TB, TB, 0, stream>>>(q, stats, g3, be3, NGRAPH, D3F);
  head_kernel<<<(NGRAPH + TB - 1) / TB, TB, 0, stream>>>(q, Wl2, bl2, out);
}
